// ResidualBlock_75797582840194
// MI455X (gfx1250) — matrix operand order repaired and checked by reading, not compiled, not run on hardware
//
#include <hip/hip_runtime.h>

// ---------------------------------------------------------------------------
// MI455X (gfx1250, wave32) sparse-conv residual block.
//  - GEMM core: v_wmma_f32_16x16x32_bf16, 2 output tiles per wave (B reuse)
//  - Weights double-buffered into LDS via global_load_async_to_lds_b128
//    (ASYNCcnt) and read back as ds_load_b128.
// ---------------------------------------------------------------------------

typedef __attribute__((ext_vector_type(16))) __bf16       v16bf;
typedef __attribute__((ext_vector_type(8)))  float        v8f;
typedef __attribute__((ext_vector_type(4)))  unsigned int v4u;

#define LRELU_NEG 0.01f
#define BN_EPS    1e-5f
#define KOFF      27
#define CCH       64
// packed-weight elements: 27 k-offsets * 2 k-steps * 4 n-tiles * 32 lanes * 16 bf16
#define KSLICE_ELEMS 4096                    // elements per k-offset (8 KB)
#define WPACK_ELEMS  (KOFF * KSLICE_ELEMS)

__device__ __forceinline__ unsigned short f2bf(float f) {
  union { float f; unsigned int u; } x; x.f = f;
  unsigned int u = x.u;
  unsigned int r = u + 0x7FFFu + ((u >> 16) & 1u);   // round-to-nearest-even
  return (unsigned short)(r >> 16);
}

// ---- feats (f32) -> bf16, plus one zero sentinel row at index n ------------
__global__ void cvt_feats_kernel(const float* __restrict__ f,
                                 unsigned short* __restrict__ fb, int n64) {
  int i = blockIdx.x * blockDim.x + threadIdx.x;
  if (i < n64)                fb[i] = f2bf(f[i]);
  else if (i < n64 + CCH)     fb[i] = 0;            // sentinel row -> zeros
}

// ---- pack W[k][cin][cout] (f32) into per-lane WMMA-B fragment layout -------
// wp[k*4096 + (s*4+nt)*512 + lane*16 + e] =
//     bf16( W[k][ s*32 + (e>>3)*16 + (lane>>4)*8 + (e&7) ][ nt*16 + (lane&15) ] )
__global__ void pack_w_kernel(const float* __restrict__ W,
                              unsigned short* __restrict__ wp) {
  int i = blockIdx.x * blockDim.x + threadIdx.x;
  if (i >= WPACK_ELEMS) return;
  int e    = i & 15;
  int lane = (i >> 4) & 31;
  int frag = i >> 9;
  int nt   = frag & 3;
  int s    = (frag >> 2) & 1;
  int k    = frag >> 3;
  int cin  = s * 32 + ((e >> 3) * 16) + ((lane >> 4) * 8) + (e & 7);   // B layout: lanes 0-15 K0-7,16-23; 16-31 K8-15,24-31
  int cout = nt * 16 + (lane & 15);
  wp[i] = f2bf(W[((size_t)k * CCH + cin) * CCH + cout]);
}

// ---- async copy of one 8KB weight slice (512 x 16B, 256 threads x 2) -------
__device__ __forceinline__ void async_copy_slice(const unsigned short* src,
                                                 unsigned int ldsoff, int tid) {
  unsigned long long g = (unsigned long long)src + (unsigned)(tid * 16);
  unsigned int       l = ldsoff + (unsigned)(tid * 16);
  asm volatile("global_load_async_to_lds_b128 %0, %1, off"
               :: "v"(l), "v"(g) : "memory");
  asm volatile("global_load_async_to_lds_b128 %0, %1, off"
               :: "v"(l + 4096u), "v"(g + 4096ull) : "memory");
}

// ---- gather + WMMA GEMM (2 tiles/wave) + per-channel sum/sumsq atomics -----
__global__ __launch_bounds__(256)
void spconv_wmma_kernel(const unsigned short* __restrict__ fin,  // (n+1) x 64 bf16
                        const unsigned short* __restrict__ wp,   // packed weights
                        const int*            __restrict__ nbr,  // 27 x n
                        float*                __restrict__ yraw, // n x 64 f32
                        float*                __restrict__ stat, // [0:64) sum, [64:128) sumsq
                        int n) {
  __shared__ unsigned short wlds[2][KSLICE_ELEMS];   // 16 KB double buffer

  const int tid  = threadIdx.x;
  const int lane = tid & 31;
  const int m    = lane & 15;    // A row within tile / B-C column within n-tile
  const int hi   = lane >> 4;
  const int tp   = (blockIdx.x * 8 + (tid >> 5)) * 2;   // tile pair base
  const int row0 = tp * 16 + m;
  const int row1 = row0 + 16;
  const bool rv0 = (row0 < n);
  const bool rv1 = (row1 < n);

  const unsigned int lds0 = (unsigned int)(uintptr_t)(&wlds[0][0]);

  // prologue: stream slices k=0 (buf0) and k=1 (buf1)
  async_copy_slice(wp,                lds0,         tid);
  async_copy_slice(wp + KSLICE_ELEMS, lds0 + 8192u, tid);

  v8f acc[2][4] = {};
  union Frag { v4u q[2]; v16bf v; };

  for (int k = 0; k < KOFF; ++k) {
    const int i0 = rv0 ? nbr[(size_t)k * n + row0] : n;   // n == zero sentinel
    const int i1 = rv1 ? nbr[(size_t)k * n + row1] : n;
    if (rv0 && (k + 1 < KOFF))
      __builtin_prefetch(nbr + (size_t)(k + 1) * n + row0, 0, 1);  // global_prefetch_b8

    // wait for slice k's async copies, then make them block-visible
    if (k < KOFF - 1) asm volatile("s_wait_asynccnt 0x2" ::: "memory");
    else              asm volatile("s_wait_asynccnt 0x0" ::: "memory");
    __syncthreads();

    const unsigned short* a0p = fin + (size_t)i0 * CCH + hi * 8;
    const unsigned short* a1p = fin + (size_t)i1 * CCH + hi * 8;
    const unsigned short* wb  = &wlds[k & 1][0];
#pragma unroll
    for (int s = 0; s < 2; ++s) {
      Frag A0, A1;
      A0.q[0] = *(const v4u*)(a0p + s * 32);        // K = s*32 + hi*8 + [0..7]
      A0.q[1] = *(const v4u*)(a0p + s * 32 + 16);   // K = s*32 + hi*8 + 16 + [0..7]
      A1.q[0] = *(const v4u*)(a1p + s * 32);
      A1.q[1] = *(const v4u*)(a1p + s * 32 + 16);
#pragma unroll
      for (int nt = 0; nt < 4; ++nt) {
        Frag B;
        const unsigned short* bl = wb + (s * 4 + nt) * 512 + lane * 16;
        B.q[0] = *(const v4u*)(bl);                 // ds_load_b128
        B.q[1] = *(const v4u*)(bl + 8);
        acc[0][nt] = __builtin_amdgcn_wmma_f32_16x16x32_bf16(
            false, A0.v, false, B.v, (short)0, acc[0][nt], false, false);
        acc[1][nt] = __builtin_amdgcn_wmma_f32_16x16x32_bf16(
            false, A1.v, false, B.v, (short)0, acc[1][nt], false, false);
      }
    }
    __syncthreads();   // everyone done reading buf[k&1] before it is refilled
    if (k + 2 < KOFF)
      async_copy_slice(wp + (size_t)(k + 2) * KSLICE_ELEMS,
                       lds0 + (unsigned)((k & 1) * 8192), tid);
  }

  // C/D layout: lane holds column (lane&15); VGPR v holds row hi*8 + v.
  const int rb0 = tp * 16 + hi * 8;
  const int rb1 = rb0 + 16;
  float* yo0 = yraw + (size_t)rb0 * CCH + m;
  float* yo1 = yraw + (size_t)rb1 * CCH + m;
#define RS(ACC, NT, RB, YO)                                                 \
  do {                                                                      \
    float cs = 0.f, cq = 0.f;                                               \
    _Pragma("unroll") for (int v = 0; v < 8; ++v) {                         \
      if ((RB) + v < n) {                                                   \
        float x = ACC[v];                                                   \
        cs += x; cq += x * x;                                               \
        (YO)[(size_t)v * CCH + (NT) * 16] = x;                              \
      }                                                                     \
    }                                                                       \
    atomicAdd(stat + (NT) * 16 + m, cs);                                    \
    atomicAdd(stat + 64 + (NT) * 16 + m, cq);                               \
  } while (0)
  RS(acc[0][0], 0, rb0, yo0); RS(acc[0][1], 1, rb0, yo0);
  RS(acc[0][2], 2, rb0, yo0); RS(acc[0][3], 3, rb0, yo0);
  RS(acc[1][0], 0, rb1, yo1); RS(acc[1][1], 1, rb1, yo1);
  RS(acc[1][2], 2, rb1, yo1); RS(acc[1][3], 3, rb1, yo1);
#undef RS
}

// ---- BN (training stats) + LeakyReLU -> bf16 activations (+ sentinel) -----
__global__ void bnact_kernel(const float* __restrict__ yraw,
                             const float* __restrict__ stat,
                             const float* __restrict__ gam,
                             const float* __restrict__ bet,
                             unsigned short* __restrict__ act, int n) {
  int i = blockIdx.x * blockDim.x + threadIdx.x;
  int total = n * CCH;
  if (i < total) {
    int   c    = i & (CCH - 1);
    float inv  = 1.0f / (float)n;
    float mean = stat[c] * inv;
    float var  = stat[64 + c] * inv - mean * mean;
    float sc   = gam[c] * rsqrtf(var + BN_EPS);
    float sh   = bet[c] - mean * sc;
    float y    = yraw[i] * sc + sh;
    y = (y >= 0.f) ? y : LRELU_NEG * y;
    act[i] = f2bf(y);
  } else if (i < total + CCH) {
    act[i] = 0;                                   // sentinel row
  }
}

// ---- BN2 + LeakyReLU + residual + LeakyReLU -> f32 output -----------------
__global__ void final_kernel(const float* __restrict__ feats,
                             const float* __restrict__ yraw,
                             const float* __restrict__ stat,
                             const float* __restrict__ gam,
                             const float* __restrict__ bet,
                             float* __restrict__ out, int n) {
  int i = blockIdx.x * blockDim.x + threadIdx.x;
  if (i >= n * CCH) return;
  int   c    = i & (CCH - 1);
  float inv  = 1.0f / (float)n;
  float mean = stat[c] * inv;
  float var  = stat[64 + c] * inv - mean * mean;
  float sc   = gam[c] * rsqrtf(var + BN_EPS);
  float sh   = bet[c] - mean * sc;
  float y    = yraw[i] * sc + sh;
  y = (y >= 0.f) ? y : LRELU_NEG * y;
  float r = feats[i] + y;
  out[i] = (r >= 0.f) ? r : LRELU_NEG * r;
}

// ---------------------------------------------------------------------------
extern "C" void kernel_launch(void* const* d_in, const int* in_sizes, int n_in,
                              void* d_out, int out_size, void* d_ws, size_t ws_size,
                              hipStream_t stream) {
  const float* feats = (const float*)d_in[0];
  const float* W1    = (const float*)d_in[1];
  const float* g1    = (const float*)d_in[2];
  const float* b1    = (const float*)d_in[3];
  const float* W2    = (const float*)d_in[4];
  const float* g2    = (const float*)d_in[5];
  const float* b2    = (const float*)d_in[6];
  const int*   nbr   = (const int*)d_in[7];

  const int n      = in_sizes[0] / CCH;        // 100000 voxels
  const int n64    = n * CCH;
  const int ntiles = (n + 15) / 16;
  const int ntp    = (ntiles + 1) / 2;         // tile pairs (2 tiles per wave)
  const int blocks = (ntp + 7) / 8;            // 8 waves per block

  // workspace carve (256B aligned): ~51.6 MB total
  auto al = [](size_t x) { return (x + 255) & ~(size_t)255; };
  char* p = (char*)d_ws;
  unsigned short* fpbf = (unsigned short*)p; p += al((size_t)(n + 1) * CCH * 2);
  unsigned short* actb = (unsigned short*)p; p += al((size_t)(n + 1) * CCH * 2);
  float*          yraw = (float*)p;          p += al((size_t)n64 * 4);           // reused by conv2
  unsigned short* wp1  = (unsigned short*)p; p += al((size_t)WPACK_ELEMS * 2);
  unsigned short* wp2  = (unsigned short*)p; p += al((size_t)WPACK_ELEMS * 2);
  float*          stat = (float*)p;          p += al(256 * 4);                   // [0:128) conv1, [128:256) conv2

  hipMemsetAsync(stat, 0, 256 * sizeof(float), stream);

  cvt_feats_kernel<<<(n64 + CCH + 255) / 256, 256, 0, stream>>>(feats, fpbf, n64);
  pack_w_kernel<<<(WPACK_ELEMS + 255) / 256, 256, 0, stream>>>(W1, wp1);
  pack_w_kernel<<<(WPACK_ELEMS + 255) / 256, 256, 0, stream>>>(W2, wp2);

  // conv1: one wave32 per 2 tiles (32 voxels), 8 waves per block
  spconv_wmma_kernel<<<blocks, 256, 0, stream>>>(fpbf, wp1, nbr, yraw, stat, n);
  bnact_kernel<<<(n64 + CCH + 255) / 256, 256, 0, stream>>>(yraw, stat, g1, b1, actb, n);

  // conv2 (raw buffer reused; bnact above has already consumed it)
  spconv_wmma_kernel<<<blocks, 256, 0, stream>>>(actb, wp2, nbr, yraw, stat + 128, n);
  final_kernel<<<(n64 + 255) / 256, 256, 0, stream>>>(feats, yraw, stat + 128, g2, b2, (float*)d_out, n);
}
